// ImprovedNeuralFactorizationMachine_4071628997270
// MI455X (gfx1250) — matrix operand order repaired and checked by reading, not compiled, not run on hardware
//
#include <hip/hip_runtime.h>

// ---------------------------------------------------------------------------
// Problem constants (fixed shapes from the reference)
// ---------------------------------------------------------------------------
#define BB   2048
#define FF   16384
#define DD   512
#define OO   4096
#define KTOP 20

typedef __attribute__((ext_vector_type(16))) __bf16 bf16x16;
typedef __attribute__((ext_vector_type(8)))  float  f32x8;
typedef __attribute__((ext_vector_type(4)))  float  f32x4;
typedef __attribute__((ext_vector_type(4)))  unsigned int u32x4;
typedef __attribute__((ext_vector_type(2)))  unsigned int u32x2;

union FragBF16 { bf16x16 v; u32x4 q[2]; };
union Pack4    { __bf16 b[4]; u32x2 u; };

// ---------------------------------------------------------------------------
// Top-K (k=20) per row. One block per row; thread t owns the strided elements
// {e*256 + t}, kept in registers; 20 argmax reductions with a claimed bitmask.
// ---------------------------------------------------------------------------
__global__ __launch_bounds__(256) void topk_kernel(const float* __restrict__ feats,
                                                   float* __restrict__ topv,
                                                   int* __restrict__ topi) {
  const int row = blockIdx.x;
  const int tid = threadIdx.x;
  const float* rp = feats + (size_t)row * FF;

  float v[64];
#pragma unroll
  for (int e = 0; e < 64; ++e) v[e] = rp[e * 256 + tid];

  unsigned long long claimed = 0ull;
  __shared__ float rv[256];
  __shared__ int   ri[256];

  for (int it = 0; it < KTOP; ++it) {
    float best = -1e30f;
    int   bidx = 0x7fffffff;
#pragma unroll
    for (int e = 0; e < 64; ++e) {
      const bool ok = ((claimed >> e) & 1ull) == 0ull;
      const float x = ok ? v[e] : -1e30f;
      if (x > best) { best = x; bidx = e * 256 + tid; }
    }
    rv[tid] = best;
    ri[tid] = bidx;
    __syncthreads();
    for (int s = 128; s > 0; s >>= 1) {
      if (tid < s) {
        const float ov = rv[tid + s];
        const int   oi = ri[tid + s];
        if (ov > rv[tid] || (ov == rv[tid] && oi < ri[tid])) { rv[tid] = ov; ri[tid] = oi; }
      }
      __syncthreads();
    }
    const int   widx = ri[0];
    const float wval = rv[0];
    if (tid == 0) {
      topv[row * KTOP + it] = wval;
      topi[row * KTOP + it] = widx;
    }
    if ((widx & 255) == tid) claimed |= (1ull << (widx >> 8));
    __syncthreads();
  }
}

// ---------------------------------------------------------------------------
// Sparse bi-interaction: bi[b,d] = 0.5*((sum_k w_k e_kd)^2 - sum_k (w_k e_kd)^2)
// One block per row, 512 threads (one per d), 20 gathered embedding rows.
// ---------------------------------------------------------------------------
__global__ __launch_bounds__(512) void bi_kernel(const float* __restrict__ topv,
                                                 const int* __restrict__ topi,
                                                 const float* __restrict__ emb,
                                                 float* __restrict__ bi) {
  __shared__ float lv[KTOP];
  __shared__ int   li[KTOP];
  const int row = blockIdx.x;
  const int tid = threadIdx.x;
  if (tid < KTOP) { lv[tid] = topv[row * KTOP + tid]; li[tid] = topi[row * KTOP + tid]; }
  __syncthreads();

  float se = 0.f, ss = 0.f;
#pragma unroll
  for (int k = 0; k < KTOP; ++k) {
    const float w  = lv[k];
    const float e  = emb[(size_t)li[k] * DD + tid];
    const float we = w * e;
    se += we;
    ss += we * we;
  }
  bi[(size_t)row * DD + tid] = 0.5f * (se * se - ss);
}

// ---------------------------------------------------------------------------
// BatchNorm column stats -> fused scale/shift (y = x*scale[c] + shift[c]).
// Thread t owns column c = bid*256+t; loads are lane-coalesced across columns.
// ---------------------------------------------------------------------------
__global__ __launch_bounds__(256) void bn_stats(const float* __restrict__ X,
                                                const float* __restrict__ gamma,
                                                const float* __restrict__ beta,
                                                float* __restrict__ scale,
                                                float* __restrict__ shift,
                                                int rows, int cols) {
  const int c = blockIdx.x * 256 + threadIdx.x;
  if (c >= cols) return;
  float s = 0.f, sq = 0.f;
  for (int r = 0; r < rows; ++r) {
    const float x = X[(size_t)r * cols + c];
    s += x;
    sq += x * x;
  }
  const float inv  = 1.0f / (float)rows;
  const float mean = s * inv;
  const float var  = sq * inv - mean * mean;
  const float sc   = gamma[c] * rsqrtf(var + 1e-5f);
  scale[c] = sc;
  shift[c] = beta[c] - mean * sc;
}

__global__ __launch_bounds__(256) void bn_apply(float* __restrict__ X,
                                                const float* __restrict__ scale,
                                                const float* __restrict__ shift,
                                                int n, int cols, int relu) {
  const int i = blockIdx.x * 256 + threadIdx.x;
  if (i >= n) return;
  const int c = i % cols;
  float y = fmaf(X[i], scale[c], shift[c]);
  if (relu) y = fmaxf(y, 0.f);
  X[i] = y;
}

// ---------------------------------------------------------------------------
// WMMA GEMM: C[M,N] = A[M,K] @ B[N,K]^T + bias[N]   (A,B,C fp32 row-major)
// fp32 fidelity via bf16 split: A=Ahi+Alo, B=Bhi+Blo,
//   C += Ahi*Bhi + Ahi*Blo + Alo*Bhi    (3x v_wmma_f32_16x16x32_bf16)
// Block tile 128x128, 8 waves (wave32), wave tile 32x64, K-step 32.
// Software pipeline: tile k+1 prefetched into VGPRs during the WMMA burst of
// tile k (RAW on the prefetch regs orders it, no extra barrier), tile k+2
// warmed into L2 with global_prefetch. LDS stride 40 bf16 (80 B) keeps b128
// loads 16-byte aligned and bank-conflict free.
// Requires M%128==0, N%128==0, K%32==0 (true for all three call sites).
// ---------------------------------------------------------------------------
#define TM 128
#define TN 128
#define TK 32
#define LSTR 40

__device__ __forceinline__ void split4(const f32x4 v, Pack4& hi, Pack4& lo) {
#pragma unroll
  for (int n = 0; n < 4; ++n) {
    const float  x = v[n];
    const __bf16 h = (__bf16)x;
    hi.b[n] = h;
    lo.b[n] = (__bf16)(x - (float)h);
  }
}

__global__ __launch_bounds__(256) void gemm_bf16x3(const float* __restrict__ A,
                                                   const float* __restrict__ Bm,
                                                   const float* __restrict__ bias,
                                                   float* __restrict__ C,
                                                   int M, int N, int K) {
  __shared__ __bf16 sAhi[TM * LSTR];
  __shared__ __bf16 sAlo[TM * LSTR];
  __shared__ __bf16 sBhi[TN * LSTR];
  __shared__ __bf16 sBlo[TN * LSTR];

  const int tid      = threadIdx.x;
  const int lane     = tid & 31;
  const int wave     = tid >> 5;
  const int m_w      = (wave & 3) * 32;   // 4 waves tile M
  const int n_w      = (wave >> 2) * 64;  // 2 waves tile N
  const int laneIdx  = lane & 15;
  const int laneHalf = lane >> 4;

  const int bm = blockIdx.y * TM;
  const int bn = blockIdx.x * TN;

  // Per-thread staging assignment: 4 float4s of A and 4 of B per k-step.
  int rowS[4], colS[4];
#pragma unroll
  for (int r = 0; r < 4; ++r) {
    const int i = tid + r * 256;
    rowS[r] = i >> 3;
    colS[r] = (i & 7) << 2;
  }
  const float* __restrict__ Abase[4];
  const float* __restrict__ Bbase[4];
#pragma unroll
  for (int r = 0; r < 4; ++r) {
    Abase[r] = A  + (size_t)(bm + rowS[r]) * K + colS[r];
    Bbase[r] = Bm + (size_t)(bn + rowS[r]) * K + colS[r];
  }

  f32x8 acc[2][4];
  const f32x8 zero = {0.f, 0.f, 0.f, 0.f, 0.f, 0.f, 0.f, 0.f};
#pragma unroll
  for (int i = 0; i < 2; ++i)
#pragma unroll
    for (int j = 0; j < 4; ++j) acc[i][j] = zero;

  // Prologue: prefetch tile k0 = 0 into registers.
  f32x4 pa[4], pb[4];
#pragma unroll
  for (int r = 0; r < 4; ++r) {
    pa[r] = *reinterpret_cast<const f32x4*>(Abase[r]);
    pb[r] = *reinterpret_cast<const f32x4*>(Bbase[r]);
  }

  for (int k0 = 0; k0 < K; k0 += TK) {
    // ---- stage current tile (hi/lo bf16 split) into LDS ----
#pragma unroll
    for (int r = 0; r < 4; ++r) {
      Pack4 hi, lo;
      split4(pa[r], hi, lo);
      *reinterpret_cast<u32x2*>(&sAhi[rowS[r] * LSTR + colS[r]]) = hi.u;
      *reinterpret_cast<u32x2*>(&sAlo[rowS[r] * LSTR + colS[r]]) = lo.u;
      split4(pb[r], hi, lo);
      *reinterpret_cast<u32x2*>(&sBhi[rowS[r] * LSTR + colS[r]]) = hi.u;
      *reinterpret_cast<u32x2*>(&sBlo[rowS[r] * LSTR + colS[r]]) = lo.u;
    }
    __syncthreads();

    // ---- issue global loads for tile k0+TK; they complete under the WMMAs ----
    const int k1 = k0 + TK;
    if (k1 < K) {
#pragma unroll
      for (int r = 0; r < 4; ++r) {
        pa[r] = *reinterpret_cast<const f32x4*>(Abase[r] + k1);
        pb[r] = *reinterpret_cast<const f32x4*>(Bbase[r] + k1);
      }
      // L2 warm-up for tile k0+2*TK (one lane-address per cacheline group).
      if (k1 + TK < K) {
#pragma unroll
        for (int r = 0; r < 4; ++r) {
          __builtin_prefetch(Abase[r] + k1 + TK, 0, 1);
          __builtin_prefetch(Bbase[r] + k1 + TK, 0, 1);
        }
      }
    }

    // ---- compute: 24 WMMAs per wave over the staged tile ----
    // A fragments (ISA 16-bit A 16x32 layout): lanes 0-15 K={0..7,16..23},
    // lanes 16-31 K={8..15,24..31}, row M = lane&15.
    FragBF16 ahi[2], alo[2];
#pragma unroll
    for (int i = 0; i < 2; ++i) {
      const int off = (m_w + 16 * i + laneIdx) * LSTR + laneHalf * 8;
      ahi[i].q[0] = *reinterpret_cast<const u32x4*>(&sAhi[off]);
      ahi[i].q[1] = *reinterpret_cast<const u32x4*>(&sAhi[off + 16]);
      alo[i].q[0] = *reinterpret_cast<const u32x4*>(&sAlo[off]);
      alo[i].q[1] = *reinterpret_cast<const u32x4*>(&sAlo[off + 16]);
    }

#pragma unroll
    for (int j = 0; j < 4; ++j) {
      // B fragments: lanes 0-15 hold K={0..7,16..23} of column N=lane, lanes 16-31
      // hold K={8..15,24..31} of column N=lane-16.
      FragBF16 bhi, blo;
      const int off = (n_w + 16 * j + laneIdx) * LSTR + laneHalf * 8;
      bhi.q[0] = *reinterpret_cast<const u32x4*>(&sBhi[off]);
      bhi.q[1] = *reinterpret_cast<const u32x4*>(&sBhi[off + 16]);
      blo.q[0] = *reinterpret_cast<const u32x4*>(&sBlo[off]);
      blo.q[1] = *reinterpret_cast<const u32x4*>(&sBlo[off + 16]);
#pragma unroll
      for (int i = 0; i < 2; ++i) {
        acc[i][j] = __builtin_amdgcn_wmma_f32_16x16x32_bf16(
            false, ahi[i].v, false, bhi.v, (short)0, acc[i][j], false, false);
        acc[i][j] = __builtin_amdgcn_wmma_f32_16x16x32_bf16(
            false, ahi[i].v, false, blo.v, (short)0, acc[i][j], false, false);
        acc[i][j] = __builtin_amdgcn_wmma_f32_16x16x32_bf16(
            false, alo[i].v, false, bhi.v, (short)0, acc[i][j], false, false);
      }
    }
    __syncthreads();
  }

  // Epilogue per ISA C/D layout: VGPR v -> M = v + 8*laneHalf, N = lane&15.
#pragma unroll
  for (int i = 0; i < 2; ++i) {
#pragma unroll
    for (int j = 0; j < 4; ++j) {
      const int col = bn + n_w + 16 * j + laneIdx;
      const float bv = bias ? bias[col] : 0.f;
#pragma unroll
      for (int v = 0; v < 8; ++v) {
        const int row = bm + m_w + 16 * i + laneHalf * 8 + v;
        C[(size_t)row * N + col] = acc[i][j][v] + bv;
      }
    }
  }
}

// ---------------------------------------------------------------------------
// output = global_bias + linear_out + interaction_out
// ---------------------------------------------------------------------------
__global__ __launch_bounds__(256) void combine_kernel(const float* __restrict__ lin,
                                                      const float* __restrict__ inter,
                                                      const float* __restrict__ gb,
                                                      float* __restrict__ out,
                                                      int n, int ocols) {
  const int i = blockIdx.x * 256 + threadIdx.x;
  if (i >= n) return;
  out[i] = gb[i % ocols] + lin[i] + inter[i];
}

// ---------------------------------------------------------------------------
// Launch
// ---------------------------------------------------------------------------
extern "C" void kernel_launch(void* const* d_in, const int* in_sizes, int n_in,
                              void* d_out, int out_size, void* d_ws, size_t ws_size,
                              hipStream_t stream) {
  (void)in_sizes; (void)n_in; (void)out_size; (void)ws_size;

  const float* sae   = (const float*)d_in[0];   // [B,F]
  const float* emb   = (const float*)d_in[1];   // [F,D]
  const float* lin_w = (const float*)d_in[2];   // [O,F]
  const float* lin_b = (const float*)d_in[3];   // [O]
  const float* gbias = (const float*)d_in[4];   // [O]
  const float* g1    = (const float*)d_in[5];   // [D]
  const float* b1    = (const float*)d_in[6];   // [D]
  const float* w1    = (const float*)d_in[7];   // [D,D]
  const float* mb1   = (const float*)d_in[8];   // [D]
  const float* g2    = (const float*)d_in[9];   // [D]
  const float* b2    = (const float*)d_in[10];  // [D]
  const float* w2    = (const float*)d_in[11];  // [O,D]
  const float* mb2   = (const float*)d_in[12];  // [O]

  // d_out = (output, linear_out, interaction_out), each [B,O] fp32.
  float* out_full  = (float*)d_out;
  float* out_lin   = out_full + (size_t)BB * OO;
  float* out_inter = out_lin  + (size_t)BB * OO;

  char* ws = (char*)d_ws;
  float* topv = (float*)ws; ws += (size_t)BB * KTOP * sizeof(float);
  int*   topi = (int*)ws;   ws += (size_t)BB * KTOP * sizeof(int);
  float* bi   = (float*)ws; ws += (size_t)BB * DD * sizeof(float);
  float* h    = (float*)ws; ws += (size_t)BB * DD * sizeof(float);
  float* sc1  = (float*)ws; ws += DD * sizeof(float);
  float* sh1  = (float*)ws; ws += DD * sizeof(float);
  float* sc2  = (float*)ws; ws += DD * sizeof(float);
  float* sh2  = (float*)ws; ws += DD * sizeof(float);

  // 1) top-20 sparsification
  topk_kernel<<<BB, 256, 0, stream>>>(sae, topv, topi);
  // 2) sparse bi-interaction pooling -> bi [B,D]
  bi_kernel<<<BB, DD, 0, stream>>>(topv, topi, emb, bi);
  // 3) BN1 (training-mode batch stats), applied in place
  bn_stats<<<(DD + 255) / 256, 256, 0, stream>>>(bi, g1, b1, sc1, sh1, BB, DD);
  bn_apply<<<((BB * DD) + 255) / 256, 256, 0, stream>>>(bi, sc1, sh1, BB * DD, DD, 0);
  // 4) h = bn1(bi) @ w1^T + b1
  gemm_bf16x3<<<dim3(DD / TN, BB / TM), 256, 0, stream>>>(bi, w1, mb1, h, BB, DD, DD);
  // 5) BN2 + ReLU, in place
  bn_stats<<<(DD + 255) / 256, 256, 0, stream>>>(h, g2, b2, sc2, sh2, BB, DD);
  bn_apply<<<((BB * DD) + 255) / 256, 256, 0, stream>>>(h, sc2, sh2, BB * DD, DD, 1);
  // 6) interaction_out = h @ w2^T + b2
  gemm_bf16x3<<<dim3(OO / TN, BB / TM), 256, 0, stream>>>(h, w2, mb2, out_inter, BB, OO, DD);
  // 7) linear_out = sae @ lin_w^T + lin_b   (dominant GEMM, K=16384)
  gemm_bf16x3<<<dim3(OO / TN, BB / TM), 256, 0, stream>>>(sae, lin_w, lin_b, out_lin, BB, OO, FF);
  // 8) output = global_bias + linear_out + interaction_out
  const int ntot = BB * OO;
  combine_kernel<<<(ntot + 255) / 256, 256, 0, stream>>>(out_lin, out_inter, gbias,
                                                         out_full, ntot, OO);
}
